// TransformerWithDynamicLoading_23261542875408
// MI455X (gfx1250) — matrix operand order repaired and checked by reading, not compiled, not run on hardware
//
#include <hip/hip_runtime.h>
#include <hip/hip_bf16.h>

// ---------------------------------------------------------------------------
// Types for CDNA5 WMMA (wave32, v_wmma_f32_16x16x32_bf16)
// ---------------------------------------------------------------------------
typedef __attribute__((ext_vector_type(16))) __bf16 bf16x16;
typedef __attribute__((ext_vector_type(8)))  float  f32x8;

union FragU { bf16x16 v; uint4 q[2]; };

__device__ __forceinline__ bf16x16 load_frag16(const unsigned short* p0,
                                               const unsigned short* p1) {
  FragU u;
  u.q[0] = *reinterpret_cast<const uint4*>(p0);
  u.q[1] = *reinterpret_cast<const uint4*>(p1);
  return u.v;
}

__device__ __forceinline__ f32x8 zero8() {
  f32x8 z = {0.f, 0.f, 0.f, 0.f, 0.f, 0.f, 0.f, 0.f};
  return z;
}

__device__ __forceinline__ f32x8 wmma_bf16(bf16x16 a, bf16x16 b, f32x8 c) {
  return __builtin_amdgcn_wmma_f32_16x16x32_bf16(false, a, false, b, (short)0,
                                                 c, false, false);
}

// round-to-nearest-even fp32 -> bf16 (values are finite here)
__device__ __forceinline__ unsigned short f2bf(float f) {
  unsigned int u = __float_as_uint(f);
  unsigned int r = u + 0x7fffu + ((u >> 16) & 1u);
  return (unsigned short)(r >> 16);
}

__device__ __forceinline__ float gelu_tanh(float x) {
  float x3 = x * x * x;
  return 0.5f * x * (1.f + tanhf(0.7978845608028654f * (x + 0.044715f * x3)));
}

// ---------------------------------------------------------------------------
// CDNA5 async global->LDS copy (ASYNCcnt-tracked, per-lane 16B)
// lds_addr: LDS-relative byte offset (low 32 bits of a generic LDS pointer)
// ---------------------------------------------------------------------------
__device__ __forceinline__ void async_ld_b128(unsigned lds_addr,
                                              const void* gaddr) {
  asm volatile("global_load_async_to_lds_b128 %0, %1, off"
               :: "v"(lds_addr), "v"((unsigned long long)(size_t)gaddr)
               : "memory");
}
__device__ __forceinline__ void wait_asynccnt0() {
  asm volatile("s_wait_asynccnt 0x0" ::: "memory");
}

// ---------------------------------------------------------------------------
// Problem constants
// ---------------------------------------------------------------------------
static constexpr int BB  = 2;
static constexpr int SS  = 2048;
static constexpr int HH  = 512;
static constexpr int NHD = 8;
static constexpr int DH  = 64;
static constexpr int TT  = BB * SS;      // 4096 tokens
static constexpr int EE  = 64;           // experts
static constexpr int FF  = 2048;         // ffn dim
static constexpr int CAP = 256;          // expert capacity
static constexpr int NSLOT = TT * 2;     // 8192 routing slots

// ---------------------------------------------------------------------------
// LayerNorm over H=512, one wave per token.  Writes bf16 (+ optional fp32).
// ---------------------------------------------------------------------------
__global__ __launch_bounds__(128) void ln_kernel(const float* __restrict__ x,
                                                 const float* __restrict__ g,
                                                 const float* __restrict__ b,
                                                 unsigned short* __restrict__ obf,
                                                 float* __restrict__ of32) {
  int wave = threadIdx.x >> 5, lane = threadIdx.x & 31;
  int t = blockIdx.x * 4 + wave;
  const float* xr = x + (size_t)t * HH;
  float vals[16];
  float s = 0.f, ss = 0.f;
#pragma unroll
  for (int j = 0; j < 16; j++) {
    float v = xr[lane + 32 * j];
    vals[j] = v; s += v; ss += v * v;
  }
#pragma unroll
  for (int m = 16; m >= 1; m >>= 1) {
    s  += __shfl_xor(s,  m);
    ss += __shfl_xor(ss, m);
  }
  float mean = s * (1.f / HH);
  float var  = ss * (1.f / HH) - mean * mean;
  float rstd = rsqrtf(var + 1e-5f);
#pragma unroll
  for (int j = 0; j < 16; j++) {
    int c = lane + 32 * j;
    float v = (vals[j] - mean) * rstd * g[c] + b[c];
    obf[(size_t)t * HH + c] = f2bf(v);
    if (of32) of32[(size_t)t * HH + c] = v;
  }
}

// ---------------------------------------------------------------------------
// Generic bf16 WMMA GEMM:  C[M,N] = A_bf16[M,K] @ W_f32[K,N] (+bias,+gelu,+resid)
// Block = 128 threads (4 waves), 64x64 tile, BK=64.  Batched via blockIdx.z.
// A tile is DMA'd global->LDS with GLOBAL_LOAD_ASYNC_TO_LDS_B128 while the
// fp32 weight tile is converted to bf16 and stored transposed [n][k] so
// B-fragments are clean b128 LDS loads.
// ---------------------------------------------------------------------------
template <bool B_OUTBF, bool B_BIAS, bool B_GELU, bool B_RES>
__global__ __launch_bounds__(128) void gemm_wmma_kernel(
    const unsigned short* __restrict__ A, const float* __restrict__ W,
    const float* __restrict__ bias, const float* __restrict__ resid,
    void* __restrict__ Cout, int M, int N, int K, long sA, long sW, long sB,
    long sC) {
  __shared__ unsigned short As[64 * 72];  // [row][k], stride 72 (16B aligned)
  __shared__ unsigned short Ws[64 * 72];  // [n][k],  transposed bf16
  int e = blockIdx.z;
  A += (long)e * sA;
  W += (long)e * sW;
  const float* bptr = B_BIAS ? (bias + (long)e * sB) : nullptr;
  int bm = blockIdx.y * 64, bn = blockIdx.x * 64;
  int tid = threadIdx.x, wave = tid >> 5, lane = tid & 31;
  int lm = lane & 15, hi = lane >> 4;
  int wm = (wave >> 1) * 32, wn = (wave & 1) * 32;

  f32x8 acc[2][2];
#pragma unroll
  for (int i = 0; i < 2; i++)
#pragma unroll
    for (int j = 0; j < 2; j++) acc[i][j] = zero8();

  for (int k0 = 0; k0 < K; k0 += 64) {
    __syncthreads();
    // A tile (64 x 64 bf16): async DMA, 4 x 16B chunks per thread
#pragma unroll
    for (int t = 0; t < 4; t++) {
      int c = tid + t * 128;
      int row = c >> 3, ch = c & 7;
      async_ld_b128((unsigned)(size_t)&As[row * 72 + ch * 8],
                    &A[(size_t)(bm + row) * K + k0 + ch * 8]);
    }
    // W tile (64 x 64 f32) -> bf16 transposed [n][k] (overlaps async DMA)
#pragma unroll
    for (int t = 0; t < 8; t++) {
      int c = tid + t * 128;
      int r = c >> 4, c4 = (c & 15) * 4;
      float4 w4 =
          *reinterpret_cast<const float4*>(&W[(size_t)(k0 + r) * N + bn + c4]);
      Ws[(c4 + 0) * 72 + r] = f2bf(w4.x);
      Ws[(c4 + 1) * 72 + r] = f2bf(w4.y);
      Ws[(c4 + 2) * 72 + r] = f2bf(w4.z);
      Ws[(c4 + 3) * 72 + r] = f2bf(w4.w);
    }
    if (k0 + 64 < K)  // hint next weight tile toward cache
      __builtin_prefetch(&W[(size_t)(k0 + 64) * N + bn + (tid & 15) * 4], 0, 0);
    wait_asynccnt0();
    __syncthreads();

    bf16x16 af[2][2], bfr[2][2];
#pragma unroll
    for (int mf = 0; mf < 2; mf++)
#pragma unroll
      for (int kk = 0; kk < 2; kk++) {
        const unsigned short* rp = &As[(wm + mf * 16 + lm) * 72 + kk * 32];
        af[mf][kk] = load_frag16(rp + hi * 8, rp + 16 + hi * 8);
      }
#pragma unroll
    for (int nf = 0; nf < 2; nf++)
#pragma unroll
      for (int kk = 0; kk < 2; kk++) {
        const unsigned short* rp = &Ws[(wn + nf * 16 + lm) * 72 + kk * 32];
        bfr[nf][kk] = load_frag16(rp + hi * 8, rp + 16 + hi * 8);
      }
#pragma unroll
    for (int mf = 0; mf < 2; mf++)
#pragma unroll
      for (int nf = 0; nf < 2; nf++) {
        acc[mf][nf] = wmma_bf16(af[mf][0], bfr[nf][0], acc[mf][nf]);
        acc[mf][nf] = wmma_bf16(af[mf][1], bfr[nf][1], acc[mf][nf]);
      }
  }

  // epilogue
#pragma unroll
  for (int mf = 0; mf < 2; mf++)
#pragma unroll
    for (int nf = 0; nf < 2; nf++) {
      int col = bn + wn + nf * 16 + lm;
      float bv = B_BIAS ? bptr[col] : 0.f;
#pragma unroll
      for (int i = 0; i < 8; i++) {
        int row = bm + wm + mf * 16 + i + hi * 8;
        float v = acc[mf][nf][i] + bv;
        if (B_GELU) v = gelu_tanh(v);
        if (B_RES) v += resid[(size_t)row * N + col];
        size_t idx = (size_t)row * N + col + (size_t)e * sC;
        if (B_OUTBF) ((unsigned short*)Cout)[idx] = f2bf(v);
        else         ((float*)Cout)[idx] = v;
      }
    }
}

// ---------------------------------------------------------------------------
// V transpose per head:  v_bf[T,H] -> vt[(b*NH+h)][d][s]  (bf16)
// ---------------------------------------------------------------------------
__global__ void vtrans_kernel(const unsigned short* __restrict__ vbf,
                              unsigned short* __restrict__ vt) {
  size_t id = (size_t)blockIdx.x * blockDim.x + threadIdx.x;
  int sp = (int)(id & 2047);
  int d  = (int)((id >> 11) & 63);
  int h  = (int)((id >> 17) & 7);
  int b  = (int)(id >> 20);
  vt[id] = vbf[((size_t)(b * SS + sp)) * HH + h * DH + d];
}

// ---------------------------------------------------------------------------
// Flash attention (no mask).  Block = 64 query rows of one (b,h); 4 waves,
// each wave owns 16 rows.  S = Q K^T via WMMA, online softmax, O += P V via
// WMMA (P relaid C->A layout through wave-private LDS; DS in-order per wave).
// ---------------------------------------------------------------------------
__global__ __launch_bounds__(128) void attn_kernel(
    const unsigned short* __restrict__ qbf, const unsigned short* __restrict__ kbf,
    const unsigned short* __restrict__ vt, unsigned short* __restrict__ obf) {
  __shared__ unsigned short Pbuf[4 * 16 * 72];
  int wave = threadIdx.x >> 5, lane = threadIdx.x & 31;
  int lm = lane & 15, hi = lane >> 4;
  int qt = blockIdx.x, h = blockIdx.y, b = blockIdx.z;
  int t0 = b * SS + qt * 64 + wave * 16;
  unsigned short* Pw = &Pbuf[wave * 16 * 72];

  const unsigned short* qrow = qbf + (size_t)(t0 + lm) * HH + h * DH;
  bf16x16 aq[2];
#pragma unroll
  for (int kd = 0; kd < 2; kd++)
    aq[kd] = load_frag16(qrow + kd * 32 + hi * 8, qrow + kd * 32 + 16 + hi * 8);

  f32x8 o[4];
#pragma unroll
  for (int d = 0; d < 4; d++) o[d] = zero8();
  float mr[8], l8[8];
#pragma unroll
  for (int i = 0; i < 8; i++) { mr[i] = -1e30f; l8[i] = 0.f; }

  const unsigned short* kbase = kbf + (size_t)b * SS * HH + h * DH;
  const unsigned short* vhead = vt + ((size_t)(b * NHD + h)) * DH * SS;

  for (int kt = 0; kt < SS / 64; kt++) {
    int n0 = kt * 64;
    f32x8 sfr[4];
#pragma unroll
    for (int nf = 0; nf < 4; nf++) {
      const unsigned short* krow = kbase + (size_t)(n0 + nf * 16 + lm) * HH;
      f32x8 acc = zero8();
#pragma unroll
      for (int kd = 0; kd < 2; kd++) {
        bf16x16 bk = load_frag16(krow + kd * 32 + hi * 8,
                                 krow + kd * 32 + 16 + hi * 8);
        acc = wmma_bf16(aq[kd], bk, acc);
      }
#pragma unroll
      for (int i = 0; i < 8; i++) sfr[nf][i] = acc[i] * 0.125f;  // dh^-0.5
    }
    float tm[8];
#pragma unroll
    for (int i = 0; i < 8; i++)
      tm[i] = fmaxf(fmaxf(sfr[0][i], sfr[1][i]), fmaxf(sfr[2][i], sfr[3][i]));
#pragma unroll
    for (int mask = 1; mask < 16; mask <<= 1)
#pragma unroll
      for (int i = 0; i < 8; i++) tm[i] = fmaxf(tm[i], __shfl_xor(tm[i], mask));

    float al[8];
#pragma unroll
    for (int i = 0; i < 8; i++) {
      float mn = fmaxf(mr[i], tm[i]);
      al[i] = __expf(mr[i] - mn);
      mr[i] = mn;
    }
#pragma unroll
    for (int d = 0; d < 4; d++)
#pragma unroll
      for (int i = 0; i < 8; i++) o[d][i] *= al[i];

    float ps[8];
#pragma unroll
    for (int i = 0; i < 8; i++) ps[i] = 0.f;
#pragma unroll
    for (int nf = 0; nf < 4; nf++)
#pragma unroll
      for (int i = 0; i < 8; i++) {
        float p = __expf(sfr[nf][i] - mr[i]);
        ps[i] += p;
        Pw[(i + 8 * hi) * 72 + nf * 16 + lm] = f2bf(p);
      }
#pragma unroll
    for (int mask = 1; mask < 16; mask <<= 1)
#pragma unroll
      for (int i = 0; i < 8; i++) ps[i] += __shfl_xor(ps[i], mask);
#pragma unroll
    for (int i = 0; i < 8; i++) l8[i] = l8[i] * al[i] + ps[i];

    bf16x16 pa[2];
    const unsigned short* prow = Pw + lm * 72;
#pragma unroll
    for (int kf = 0; kf < 2; kf++)
      pa[kf] = load_frag16(prow + kf * 32 + hi * 8, prow + kf * 32 + 16 + hi * 8);

#pragma unroll
    for (int df = 0; df < 4; df++) {
      const unsigned short* vrow = vhead + (size_t)(df * 16 + lm) * SS + n0;
#pragma unroll
      for (int kf = 0; kf < 2; kf++) {
        bf16x16 bv = load_frag16(vrow + kf * 32 + hi * 8,
                                 vrow + kf * 32 + 16 + hi * 8);
        o[df] = wmma_bf16(pa[kf], bv, o[df]);
      }
    }
  }
#pragma unroll
  for (int df = 0; df < 4; df++)
#pragma unroll
    for (int i = 0; i < 8; i++) {
      float v = o[df][i] / l8[i];
      obf[(size_t)(t0 + i + 8 * hi) * HH + h * DH + df * 16 + lm] = f2bf(v);
    }
}

// ---------------------------------------------------------------------------
// Router in fp32 (ranking must be numerically faithful).  Block per token.
// ---------------------------------------------------------------------------
__global__ __launch_bounds__(64) void router_kernel(const float* __restrict__ x2f,
                                                    const float* __restrict__ wr,
                                                    float* __restrict__ logits) {
  __shared__ float xs[HH];
  int t = blockIdx.x;
  for (int j = threadIdx.x; j < HH; j += 64) xs[j] = x2f[(size_t)t * HH + j];
  __syncthreads();
  int e = threadIdx.x;
  float s = 0.f;
#pragma unroll 4
  for (int k = 0; k < HH; k++) s = fmaf(xs[k], wr[k * EE + e], s);
  logits[(size_t)t * EE + e] = s;
}

__global__ void top2_kernel(const float* __restrict__ logits,
                            int* __restrict__ ef, float* __restrict__ gf) {
  int t = blockIdx.x * blockDim.x + threadIdx.x;
  if (t >= TT) return;
  const float* L = logits + (size_t)t * EE;
  float v1 = -1e30f, v2 = -1e30f;
  int i1 = 0, i2 = 0;
  for (int j = 0; j < EE; j++) {
    float f = L[j];
    if (f > v1)      { v2 = v1; i2 = i1; v1 = f; i1 = j; }
    else if (f > v2) { v2 = f; i2 = j; }
  }
  float g1 = 1.f / (1.f + __expf(v2 - v1));  // softmax-top2 renormalized
  ef[2 * t] = i1; ef[2 * t + 1] = i2;
  gf[2 * t] = g1; gf[2 * t + 1] = 1.f - g1;
}

// ---------------------------------------------------------------------------
// Deterministic slot-ordered capacity scan (matches jnp.cumsum over slots).
// ---------------------------------------------------------------------------
__global__ void hist_rank_kernel(const int* __restrict__ ef,
                                 int* __restrict__ rank, int* __restrict__ hist) {
  __shared__ int sef[256];
  __shared__ int sh[EE];
  int tid = threadIdx.x;
  if (tid < EE) sh[tid] = 0;
  __syncthreads();
  int s = blockIdx.x * 256 + tid;
  int e = ef[s];
  sef[tid] = e;
  atomicAdd(&sh[e], 1);
  __syncthreads();
  int r = 0;
  for (int j = 0; j < tid; j++) r += (sef[j] == e);
  rank[s] = r;
  if (tid < EE) hist[blockIdx.x * EE + tid] = sh[tid];
}

__global__ void scan_kernel(const int* __restrict__ hist, int* __restrict__ off) {
  int e = threadIdx.x;
  int run = 0;
  for (int b = 0; b < NSLOT / 256; b++) {
    off[b * EE + e] = run;
    run += hist[b * EE + e];
  }
}

__global__ void posk_kernel(const int* __restrict__ ef, const int* __restrict__ rank,
                            const int* __restrict__ off, const float* __restrict__ gf,
                            int* __restrict__ posk, float* __restrict__ wkeep) {
  int s = blockIdx.x * blockDim.x + threadIdx.x;
  if (s >= NSLOT) return;
  int e = ef[s];
  int p = off[(s >> 8) * EE + e] + rank[s];
  if (p < CAP) { posk[s] = p;  wkeep[s] = gf[s]; }
  else         { posk[s] = -1; wkeep[s] = 0.f;  }
}

__global__ __launch_bounds__(64) void dispatch_kernel(
    const unsigned short* __restrict__ x2bf, const int* __restrict__ ef,
    const int* __restrict__ posk, unsigned short* __restrict__ buf) {
  int s = blockIdx.x;
  int p = posk[s];
  if (p < 0) return;  // uniform per block
  int t = s >> 1, e = ef[s];
  const uint4* src = reinterpret_cast<const uint4*>(x2bf + (size_t)t * HH);
  uint4* dst = reinterpret_cast<uint4*>(buf + ((size_t)e * CAP + p) * HH);
  dst[threadIdx.x] = src[threadIdx.x];  // 64 * 16B = 512 bf16
}

// ---------------------------------------------------------------------------
// Combine:  out = x_attn + sum_k wkeep_k * y[e_k][pos_k]
// ---------------------------------------------------------------------------
__global__ __launch_bounds__(128) void combine_kernel(
    const float* __restrict__ xattn, const float* __restrict__ y,
    const int* __restrict__ ef, const int* __restrict__ posk,
    const float* __restrict__ wk, float* __restrict__ out) {
  int t = blockIdx.x;
  int s0 = 2 * t, s1 = 2 * t + 1;
  int e0 = ef[s0], e1 = ef[s1];
  int p0 = posk[s0], p1 = posk[s1];
  float w0 = wk[s0], w1 = wk[s1];
  const float* y0 = y + ((size_t)e0 * CAP + (p0 < 0 ? 0 : p0)) * HH;
  const float* y1 = y + ((size_t)e1 * CAP + (p1 < 0 ? 0 : p1)) * HH;
#pragma unroll
  for (int j = 0; j < 4; j++) {
    int c = threadIdx.x + j * 128;
    out[(size_t)t * HH + c] =
        xattn[(size_t)t * HH + c] + w0 * y0[c] + w1 * y1[c];
  }
}

// ---------------------------------------------------------------------------
// Host launcher
// ---------------------------------------------------------------------------
extern "C" void kernel_launch(void* const* d_in, const int* in_sizes, int n_in,
                              void* d_out, int out_size, void* d_ws,
                              size_t ws_size, hipStream_t stream) {
  (void)in_sizes; (void)n_in; (void)out_size; (void)ws_size;
  const float* x    = (const float*)d_in[0];
  const float* ln1g = (const float*)d_in[1];
  const float* ln1b = (const float*)d_in[2];
  const float* wq   = (const float*)d_in[3];
  const float* wk   = (const float*)d_in[4];
  const float* wv   = (const float*)d_in[5];
  const float* wo   = (const float*)d_in[6];
  const float* ln2g = (const float*)d_in[7];
  const float* ln2b = (const float*)d_in[8];
  const float* wr   = (const float*)d_in[9];
  const float* w1   = (const float*)d_in[10];
  const float* b1   = (const float*)d_in[11];
  const float* w2   = (const float*)d_in[12];
  const float* b2   = (const float*)d_in[13];
  float* out = (float*)d_out;

  char* ws = (char*)d_ws;
  size_t off = 0;
  auto take = [&](size_t bytes) {
    size_t c = off;
    off += (bytes + 255) & ~(size_t)255;
    return c;
  };
  unsigned short* x1bf  = (unsigned short*)(ws + take((size_t)TT * HH * 2));
  unsigned short* qbf   = (unsigned short*)(ws + take((size_t)TT * HH * 2));
  unsigned short* kbf   = (unsigned short*)(ws + take((size_t)TT * HH * 2));
  unsigned short* vbf   = (unsigned short*)(ws + take((size_t)TT * HH * 2));
  unsigned short* vtbf  = (unsigned short*)(ws + take((size_t)TT * HH * 2));
  unsigned short* atbf  = (unsigned short*)(ws + take((size_t)TT * HH * 2));
  float*          xattn = (float*)(ws + take((size_t)TT * HH * 4));
  unsigned short* x2bf  = (unsigned short*)(ws + take((size_t)TT * HH * 2));
  float*          x2f   = (float*)(ws + take((size_t)TT * HH * 4));
  float*          logit = (float*)(ws + take((size_t)TT * EE * 4));
  int*            ef    = (int*)(ws + take((size_t)NSLOT * 4));
  float*          gf    = (float*)(ws + take((size_t)NSLOT * 4));
  int*            rankA = (int*)(ws + take((size_t)NSLOT * 4));
  int*            hist  = (int*)(ws + take((size_t)(NSLOT / 256) * EE * 4));
  int*            offA  = (int*)(ws + take((size_t)(NSLOT / 256) * EE * 4));
  int*            posk  = (int*)(ws + take((size_t)NSLOT * 4));
  float*          wkeep = (float*)(ws + take((size_t)NSLOT * 4));
  unsigned short* buf   = (unsigned short*)(ws + take((size_t)EE * CAP * HH * 2));
  unsigned short* hbf   = (unsigned short*)(ws + take((size_t)EE * CAP * FF * 2));
  float*          yexp  = (float*)(ws + take((size_t)EE * CAP * HH * 4));

  dim3 blk128(128);

  // 1) LN1 -> bf16
  ln_kernel<<<dim3(TT / 4), blk128, 0, stream>>>(x, ln1g, ln1b, x1bf, nullptr);

  // 2) QKV projections (bf16 WMMA, bf16 out)
  dim3 gQKV(HH / 64, TT / 64, 1);
  gemm_wmma_kernel<true, false, false, false><<<gQKV, blk128, 0, stream>>>(
      x1bf, wq, nullptr, nullptr, qbf, TT, HH, HH, 0, 0, 0, 0);
  gemm_wmma_kernel<true, false, false, false><<<gQKV, blk128, 0, stream>>>(
      x1bf, wk, nullptr, nullptr, kbf, TT, HH, HH, 0, 0, 0, 0);
  gemm_wmma_kernel<true, false, false, false><<<gQKV, blk128, 0, stream>>>(
      x1bf, wv, nullptr, nullptr, vbf, TT, HH, HH, 0, 0, 0, 0);

  // 3) V transpose per head
  vtrans_kernel<<<dim3((TT * HH) / 256), dim3(256), 0, stream>>>(vbf, vtbf);

  // 4) Flash attention
  attn_kernel<<<dim3(SS / 64, NHD, BB), blk128, 0, stream>>>(qbf, kbf, vtbf, atbf);

  // 5) WO projection + residual (fp32 out)
  gemm_wmma_kernel<false, false, false, true><<<gQKV, blk128, 0, stream>>>(
      atbf, wo, nullptr, x, xattn, TT, HH, HH, 0, 0, 0, 0);

  // 6) LN2 -> bf16 + fp32
  ln_kernel<<<dim3(TT / 4), blk128, 0, stream>>>(xattn, ln2g, ln2b, x2bf, x2f);

  // 7) Router (fp32), top-2, capacity scan
  router_kernel<<<dim3(TT), dim3(64), 0, stream>>>(x2f, wr, logit);
  top2_kernel<<<dim3(TT / 256), dim3(256), 0, stream>>>(logit, ef, gf);
  hist_rank_kernel<<<dim3(NSLOT / 256), dim3(256), 0, stream>>>(ef, rankA, hist);
  scan_kernel<<<dim3(1), dim3(EE), 0, stream>>>(hist, offA);
  posk_kernel<<<dim3(NSLOT / 256), dim3(256), 0, stream>>>(ef, rankA, offA, gf,
                                                           posk, wkeep);

  // 8) Dispatch into expert buffers
  hipMemsetAsync(buf, 0, (size_t)EE * CAP * HH * 2, stream);
  dispatch_kernel<<<dim3(NSLOT), dim3(64), 0, stream>>>(x2bf, ef, posk, buf);

  // 9) Expert FFN (batched over experts)
  dim3 gF1(FF / 64, CAP / 64, EE);
  gemm_wmma_kernel<true, true, true, false><<<gF1, blk128, 0, stream>>>(
      buf, w1, b1, nullptr, hbf, CAP, FF, HH, (long)CAP * HH, (long)HH * FF,
      (long)FF, (long)CAP * FF);
  dim3 gF2(HH / 64, CAP / 64, EE);
  gemm_wmma_kernel<false, true, false, false><<<gF2, blk128, 0, stream>>>(
      hbf, w2, b2, nullptr, yexp, CAP, HH, FF, (long)CAP * FF, (long)FF * HH,
      (long)HH, (long)CAP * HH);

  // 10) Combine + final residual
  combine_kernel<<<dim3(TT), blk128, 0, stream>>>(xattn, yexp, ef, posk, wkeep,
                                                  out);
}
